// DGCNN_39384850104584
// MI455X (gfx1250) — matrix operand order repaired and checked by reading, not compiled, not run on hardware
//
#include <hip/hip_runtime.h>
#include <hip/hip_bf16.h>
#include <math.h>

// ---------------------------------------------------------------------------
// DGCNN for MI455X (gfx1250): edge-MLP GEMMs on v_wmma_f32_16x16x32_f16,
// 32x32 output tile per wave (4 WMMA per k-step, 2x operand reuse),
// global_prefetch on the streaming A operand; f32 VALU for kNN + head.
// ---------------------------------------------------------------------------

#define BD   16      // batch (clouds)
#define ND   1024    // points per cloud
#define KNN  20      // neighbors
#define MB   (ND * KNN)   // 20480 edge rows per cloud
#define CHT  512     // concat channels (64+64+128+256)

typedef __attribute__((ext_vector_type(16))) _Float16 v16h;
typedef __attribute__((ext_vector_type(8)))  _Float16 v8h;
typedef __attribute__((ext_vector_type(8)))  float    v8f;

// ------------------------------- weight pack -------------------------------
// Pack f32 weight W[Kreal, Nout] (K-padded to Kpad, zero fill) into this file's contiguous-k
// WMMA B-operand tile layout:  offset = ((kt*Ntiles + nt)*32 + lane)*16 + i
//   lanes 0-15  hold K = kt*32 + 0..15  for column n = nt*16 + lane
//   lanes 16-31 hold K = kt*32 + 16..31 for column n = nt*16 + (lane-16)
__global__ void pack_weight_kernel(const float* __restrict__ W,
                                   _Float16* __restrict__ Bp,
                                   int Kreal, int Kpad, int Nout) {
    int t = blockIdx.x * blockDim.x + threadIdx.x;
    int total = Kpad * Nout;
    if (t >= total) return;
    int n = t % Nout;
    int k = t / Nout;
    float v = (k < Kreal) ? W[(size_t)k * Nout + n] : 0.0f;
    int kt = k >> 5, kk = k & 31;
    int nt = n >> 4, nc = n & 15;
    int lane = (kk < 16) ? nc : (16 + nc);
    int i    = (kk < 16) ? kk : (kk - 16);
    int Ntiles = Nout >> 4;
    size_t off = (((size_t)kt * Ntiles + nt) * 32 + lane) * 16 + i;
    Bp[off] = (_Float16)v;
}

// ------------------------------ kNN pipeline -------------------------------
__global__ void sqnorm_kernel(const float* __restrict__ x, int ldx, int C,
                              float* __restrict__ sq) {
    int n = blockIdx.x * blockDim.x + threadIdx.x;
    if (n >= ND) return;
    const float* p = x + (size_t)n * ldx;
    float s = 0.0f;
    for (int c = 0; c < C; ++c) s = fmaf(p[c], p[c], s);
    sq[n] = s;
}

__global__ void dist_kernel(const float* __restrict__ x, int ldx, int C,
                            const float* __restrict__ sq,
                            float* __restrict__ D) {
    int t = blockIdx.x * blockDim.x + threadIdx.x;
    if (t >= ND * ND) return;
    int n = t >> 10;          // /ND
    int m = t & (ND - 1);     // %ND
    const float* xi = x + (size_t)n * ldx;
    const float* xj = x + (size_t)m * ldx;
    float dot = 0.0f;
    for (int c = 0; c < C; ++c) dot = fmaf(xi[c], xj[c], dot);
    D[t] = sq[n] + sq[m] - 2.0f * dot;
}

// register-resident unrolled top-KNN (smallest distance) per row
__global__ void topk_kernel(const float* __restrict__ D,
                            int* __restrict__ idx_out) {
    int n = blockIdx.x * blockDim.x + threadIdx.x;
    if (n >= ND) return;
    float best[KNN];
    int   bidx[KNN];
#pragma unroll
    for (int k = 0; k < KNN; ++k) { best[k] = 3.0e38f; bidx[k] = 0; }
    const float* row = D + (size_t)n * ND;
    for (int m = 0; m < ND; ++m) {
        float d = row[m];
        if (d < best[KNN - 1]) {
            int mi = m;
#pragma unroll
            for (int p = 0; p < KNN; ++p) {
                if (d < best[p]) {
                    float td = best[p]; best[p] = d; d = td;
                    int   ti = bidx[p]; bidx[p] = mi; mi = ti;
                }
            }
        }
    }
#pragma unroll
    for (int k = 0; k < KNN; ++k) idx_out[n * KNN + k] = bidx[k];
}

// edge feature rows: E[r = n*KNN+k][c] = [x_i , x_j - x_i , 0-pad] in f16
__global__ void gather_kernel(const float* __restrict__ x, int ldx, int C,
                              const int* __restrict__ idx,
                              _Float16* __restrict__ E, int Kpad) {
    size_t t = (size_t)blockIdx.x * blockDim.x + threadIdx.x;
    size_t total = (size_t)MB * Kpad;
    if (t >= total) return;
    int c = (int)(t % Kpad);
    size_t r = t / Kpad;
    int k = (int)(r % KNN);
    int n = (int)(r / KNN);
    float v = 0.0f;
    if (c < C) {
        v = x[(size_t)n * ldx + c];
    } else if (c < 2 * C) {
        int cc = c - C;
        int j = idx[n * KNN + k];
        v = x[(size_t)j * ldx + cc] - x[(size_t)n * ldx + cc];
    }
    E[t] = (_Float16)v;
}

// ------------------------------- WMMA GEMM ---------------------------------
// Out[M,Nout] = act(A[M,Kpad] @ W + bias).  A row-major f16; W pre-packed.
// One wave -> 32x32 output tile: 2 M-fragments x 2 N-fragments share loads,
// 4 v_wmma per k-step.  Tile id is wave-uniform so EXEC stays all-ones
// around v_wmma (ISA requirement).  M % 32 == 0, Ntiles even by design.
__global__ __launch_bounds__(128)
void wmma_gemm_kernel(const _Float16* __restrict__ A,
                      const _Float16* __restrict__ Bp,
                      const float* __restrict__ bias,
                      void* __restrict__ Out,
                      int M, int Kpad, int Nout, int relu, int out_f16) {
    const int lane = threadIdx.x & 31;
    const int wv   = threadIdx.x >> 5;
    const int Ntiles = Nout >> 4;
    const int Ktiles = Kpad >> 5;
    const int MT2 = M >> 5;                    // 32-row groups
    const int NT2 = Ntiles >> 1;               // 32-col groups
    int quad = blockIdx.x * 4 + wv;
    if (quad >= MT2 * NT2) return;             // whole-wave uniform exit
    int mt2 = quad / NT2;
    int nt  = (quad % NT2) * 2;

    const int hs  = lane >> 4;                 // half-wave select
    const int l16 = lane & 15;
    // A fragment (16-bit 16x32): lane row fixed;
    //   hs=0 -> halfs {K 0..15}; hs=1 -> {K 16..31}
    const _Float16* arow0 = A + (size_t)(mt2 * 32 + l16) * Kpad + hs * 16;
    const _Float16* arow1 = arow0 + (size_t)16 * Kpad;

    union V16 { v16h v; v8h h[2]; };
    v8f acc00 = {}, acc01 = {}, acc10 = {}, acc11 = {};
    for (int kt = 0; kt < Ktiles; ++kt) {
        // prefetch the A fragments two k-steps ahead (speculative, OOB-safe)
        __builtin_prefetch(arow0 + (kt + 2) * 32, 0, 1);
        __builtin_prefetch(arow1 + (kt + 2) * 32, 0, 1);
        V16 a0, a1;
        a0.h[0] = *(const v8h*)(arow0 + kt * 32);
        a0.h[1] = *(const v8h*)(arow0 + kt * 32 + 8);
        a1.h[0] = *(const v8h*)(arow1 + kt * 32);
        a1.h[1] = *(const v8h*)(arow1 + kt * 32 + 8);
        const _Float16* bb = Bp + (((size_t)kt * Ntiles + nt) * 32 + lane) * 16;
        v16h b0 = *(const v16h*)(bb);
        v16h b1 = *(const v16h*)(bb + 512);    // next 16x32 B tile
        acc00 = __builtin_amdgcn_wmma_f32_16x16x32_f16(
            false, a0.v, false, b0, (short)0, acc00, false, false);
        acc01 = __builtin_amdgcn_wmma_f32_16x16x32_f16(
            false, a0.v, false, b1, (short)0, acc01, false, false);
        acc10 = __builtin_amdgcn_wmma_f32_16x16x32_f16(
            false, a1.v, false, b0, (short)0, acc10, false, false);
        acc11 = __builtin_amdgcn_wmma_f32_16x16x32_f16(
            false, a1.v, false, b1, (short)0, acc11, false, false);
    }

    // C/D layout: VGPR r -> row tileRow + r + hs*8, col tileCol + l16
    int col0 = nt * 16 + l16;
    int col1 = col0 + 16;
    float bi0 = bias[col0];
    float bi1 = bias[col1];
#pragma unroll
    for (int r = 0; r < 8; ++r) {
        int row0 = mt2 * 32 + r + hs * 8;
        int row1 = row0 + 16;
        float v00 = acc00[r] + bi0;
        float v01 = acc01[r] + bi1;
        float v10 = acc10[r] + bi0;
        float v11 = acc11[r] + bi1;
        if (relu) {
            v00 = fmaxf(v00, 0.0f); v01 = fmaxf(v01, 0.0f);
            v10 = fmaxf(v10, 0.0f); v11 = fmaxf(v11, 0.0f);
        }
        if (out_f16) {
            _Float16* o0 = (_Float16*)Out + (size_t)row0 * Nout;
            _Float16* o1 = (_Float16*)Out + (size_t)row1 * Nout;
            o0[col0] = (_Float16)v00; o0[col1] = (_Float16)v01;
            o1[col0] = (_Float16)v10; o1[col1] = (_Float16)v11;
        } else {
            float* o0 = (float*)Out + (size_t)row0 * Nout;
            float* o1 = (float*)Out + (size_t)row1 * Nout;
            o0[col0] = v00; o0[col1] = v01;
            o1[col0] = v10; o1[col1] = v11;
        }
    }
}

// max over K neighbors -> write channel slice of XF[b] (row stride CHT)
__global__ void maxk_kernel(const float* __restrict__ EO, int Co,
                            float* __restrict__ xout) {
    int t = blockIdx.x * blockDim.x + threadIdx.x;
    if (t >= ND * Co) return;
    int c = t % Co;
    int n = t / Co;
    const float* p = EO + ((size_t)n * KNN) * Co + c;
    float m = p[0];
#pragma unroll
    for (int k = 1; k < KNN; ++k) m = fmaxf(m, p[(size_t)k * Co]);
    xout[(size_t)n * CHT + c] = m;
}

// --------------------------------- head ------------------------------------
__global__ void pool_kernel(const float* __restrict__ XF,
                            float* __restrict__ pooled) {
    int t = blockIdx.x * blockDim.x + threadIdx.x;
    if (t >= BD * CHT) return;
    int c = t % CHT;
    int b = t / CHT;
    const float* p = XF + ((size_t)b * ND) * CHT + c;
    float m = p[0];
    for (int n = 1; n < ND; ++n) m = fmaxf(m, p[(size_t)n * CHT]);
    pooled[t] = m;
}

__global__ void lin1_kernel(const float* __restrict__ pooled,
                            const float* __restrict__ W,   // [512,1024]
                            const float* __restrict__ bias,
                            float* __restrict__ H) {
    int t = blockIdx.x * blockDim.x + threadIdx.x;
    if (t >= BD * 1024) return;
    int o = t % 1024;
    int b = t / 1024;
    const float* p = pooled + (size_t)b * CHT;
    float s = bias[o];
    for (int c = 0; c < CHT; ++c) s = fmaf(p[c], W[(size_t)c * 1024 + o], s);
    H[t] = s;
}

__global__ void bnstats_kernel(const float* __restrict__ H,
                               float* __restrict__ mu,
                               float* __restrict__ var) {
    int o = blockIdx.x * blockDim.x + threadIdx.x;
    if (o >= 1024) return;
    float m = 0.0f;
    for (int b = 0; b < BD; ++b) m += H[(size_t)b * 1024 + o];
    m *= (1.0f / BD);
    float v = 0.0f;
    for (int b = 0; b < BD; ++b) {
        float d = H[(size_t)b * 1024 + o] - m;
        v = fmaf(d, d, v);
    }
    mu[o] = m;
    var[o] = v * (1.0f / BD);
}

__global__ void bnrelu_kernel(float* __restrict__ H,
                              const float* __restrict__ mu,
                              const float* __restrict__ var,
                              const float* __restrict__ gamma,
                              const float* __restrict__ beta) {
    int t = blockIdx.x * blockDim.x + threadIdx.x;
    if (t >= BD * 1024) return;
    int o = t % 1024;
    float h = (H[t] - mu[o]) * rsqrtf(var[o] + 1e-5f) * gamma[o] + beta[o];
    H[t] = fmaxf(h, 0.0f);
}

__global__ void lin2_kernel(const float* __restrict__ H,
                            const float* __restrict__ W,   // [1024,40]
                            const float* __restrict__ bias,
                            float* __restrict__ logits) {
    int t = blockIdx.x * blockDim.x + threadIdx.x;
    if (t >= BD * 40) return;
    int o = t % 40;
    int b = t / 40;
    const float* p = H + (size_t)b * 1024;
    float s = bias[o];
    for (int c = 0; c < 1024; ++c) s = fmaf(p[c], W[(size_t)c * 40 + o], s);
    logits[t] = s;
}

__global__ void logsoftmax_kernel(const float* __restrict__ logits,
                                  float* __restrict__ out) {
    int b = blockIdx.x * blockDim.x + threadIdx.x;
    if (b >= BD) return;
    const float* r = logits + b * 40;
    float mx = r[0];
    for (int i = 1; i < 40; ++i) mx = fmaxf(mx, r[i]);
    float s = 0.0f;
    for (int i = 0; i < 40; ++i) s += expf(r[i] - mx);
    float ls = logf(s);
    for (int i = 0; i < 40; ++i) out[b * 40 + i] = r[i] - mx - ls;
}

// ------------------------------- launcher ----------------------------------
static inline int ceil_div(long long a, int b) { return (int)((a + b - 1) / b); }

extern "C" void kernel_launch(void* const* d_in, const int* in_sizes, int n_in,
                              void* d_out, int out_size, void* d_ws, size_t ws_size,
                              hipStream_t stream) {
    (void)in_sizes; (void)n_in; (void)out_size; (void)ws_size;
    const float* pos    = (const float*)d_in[0];
    // d_in[1] = batch (int64) unused: equal-sized clouds
    const float* W1a = (const float*)d_in[2];  const float* b1a = (const float*)d_in[3];
    const float* W1b = (const float*)d_in[4];  const float* b1b = (const float*)d_in[5];
    const float* W2a = (const float*)d_in[6];  const float* b2a = (const float*)d_in[7];
    const float* W2b = (const float*)d_in[8];  const float* b2b = (const float*)d_in[9];
    const float* W3a = (const float*)d_in[10]; const float* b3a = (const float*)d_in[11];
    const float* W3b = (const float*)d_in[12]; const float* b3b = (const float*)d_in[13];
    const float* W4a = (const float*)d_in[14]; const float* b4a = (const float*)d_in[15];
    const float* W4b = (const float*)d_in[16]; const float* b4b = (const float*)d_in[17];
    const float* lin1_w = (const float*)d_in[18]; const float* lin1_b = (const float*)d_in[19];
    const float* gamma  = (const float*)d_in[20]; const float* beta   = (const float*)d_in[21];
    const float* lin2_w = (const float*)d_in[22]; const float* lin2_b = (const float*)d_in[23];
    float* out = (float*)d_out;

    // -------- workspace carve-out (deterministic, no allocation) --------
    char* cur = (char*)d_ws;
    auto take = [&](size_t bytes) -> void* {
        void* p = (void*)cur;
        cur += (bytes + 255) & ~(size_t)255;
        return p;
    };
    float*    XF   = (float*)take((size_t)BD * ND * CHT * sizeof(float));   // 33.5 MB
    int*      IDX  = (int*)take((size_t)BD * ND * KNN * sizeof(int));
    float*    SQ   = (float*)take((size_t)ND * sizeof(float));
    float*    D    = (float*)take((size_t)ND * ND * sizeof(float));          // 4 MB
    _Float16* E    = (_Float16*)take((size_t)MB * 512 * sizeof(_Float16));   // 21 MB
    _Float16* Hh   = (_Float16*)take((size_t)MB * 256 * sizeof(_Float16));   // 10.5 MB
    float*    EO   = (float*)take((size_t)MB * 256 * sizeof(float));         // 21 MB
    _Float16* P1a  = (_Float16*)take((size_t)32  * 64  * sizeof(_Float16));
    _Float16* P1b  = (_Float16*)take((size_t)64  * 64  * sizeof(_Float16));
    _Float16* P2a  = (_Float16*)take((size_t)128 * 64  * sizeof(_Float16));
    _Float16* P2b  = (_Float16*)take((size_t)64  * 64  * sizeof(_Float16));
    _Float16* P3a  = (_Float16*)take((size_t)128 * 128 * sizeof(_Float16));
    _Float16* P3b  = (_Float16*)take((size_t)128 * 128 * sizeof(_Float16));
    _Float16* P4a  = (_Float16*)take((size_t)256 * 256 * sizeof(_Float16));
    _Float16* P4b  = (_Float16*)take((size_t)256 * 256 * sizeof(_Float16));
    float*    POOL = (float*)take((size_t)BD * CHT * sizeof(float));
    float*    H1   = (float*)take((size_t)BD * 1024 * sizeof(float));
    float*    MU   = (float*)take((size_t)1024 * sizeof(float));
    float*    VAR  = (float*)take((size_t)1024 * sizeof(float));
    float*    LOG  = (float*)take((size_t)BD * 40 * sizeof(float));

    // -------- pack all MLP weights into WMMA B layout --------
    struct PackJob { const float* W; _Float16* P; int Kreal, Kpad, Nout; };
    PackJob packs[8] = {
        {W1a, P1a,   6,  32,  64}, {W1b, P1b,  64,  64,  64},
        {W2a, P2a, 128, 128,  64}, {W2b, P2b,  64,  64,  64},
        {W3a, P3a, 128, 128, 128}, {W3b, P3b, 128, 128, 128},
        {W4a, P4a, 256, 256, 256}, {W4b, P4b, 256, 256, 256},
    };
    for (int i = 0; i < 8; ++i) {
        int total = packs[i].Kpad * packs[i].Nout;
        pack_weight_kernel<<<ceil_div(total, 256), 256, 0, stream>>>(
            packs[i].W, packs[i].P, packs[i].Kreal, packs[i].Kpad, packs[i].Nout);
    }

    // -------- edge-conv layers, chunked per cloud --------
    struct LayerDef {
        int C, Kpad, Ch, Co, outOff, inOff;
        _Float16 *Pa, *Pb;
        const float *ba, *bb;
    };
    LayerDef L[4] = {
        {  3,  32,  64,  64,   0,   0, P1a, P1b, b1a, b1b},
        { 64, 128,  64,  64,  64,   0, P2a, P2b, b2a, b2b},
        { 64, 128, 128, 128, 128,  64, P3a, P3b, b3a, b3b},
        {128, 256, 256, 256, 256, 128, P4a, P4b, b4a, b4b},
    };

    for (int layer = 0; layer < 4; ++layer) {
        const LayerDef& ld = L[layer];
        for (int b = 0; b < BD; ++b) {
            const float* xin;
            int ldx;
            if (layer == 0) { xin = pos + (size_t)b * ND * 3; ldx = 3; }
            else { xin = XF + ((size_t)b * ND) * CHT + ld.inOff; ldx = CHT; }
            int* idx_b = IDX + (size_t)b * ND * KNN;

            sqnorm_kernel<<<ceil_div(ND, 256), 256, 0, stream>>>(xin, ldx, ld.C, SQ);
            dist_kernel<<<ceil_div((long long)ND * ND, 256), 256, 0, stream>>>(
                xin, ldx, ld.C, SQ, D);
            topk_kernel<<<ceil_div(ND, 128), 128, 0, stream>>>(D, idx_b);
            gather_kernel<<<ceil_div((long long)MB * ld.Kpad, 256), 256, 0, stream>>>(
                xin, ldx, ld.C, idx_b, E, ld.Kpad);

            // GEMM1: H = relu(E @ Wa + ba)   [MB, Kpad] x [Kpad, Ch] -> f16
            {
                int quads = (MB / 32) * (ld.Ch / 32);
                wmma_gemm_kernel<<<ceil_div(quads, 4), 128, 0, stream>>>(
                    E, ld.Pa, ld.ba, Hh, MB, ld.Kpad, ld.Ch, 1, 1);
            }
            // GEMM2: EO = H @ Wb + bb        [MB, Ch] x [Ch, Co] -> f32
            {
                int quads = (MB / 32) * (ld.Co / 32);
                wmma_gemm_kernel<<<ceil_div(quads, 4), 128, 0, stream>>>(
                    Hh, ld.Pb, ld.bb, EO, MB, ld.Ch, ld.Co, 0, 0);
            }
            // max over K neighbors -> XF channel slice
            maxk_kernel<<<ceil_div(ND * ld.Co, 256), 256, 0, stream>>>(
                EO, ld.Co, XF + ((size_t)b * ND) * CHT + ld.outOff);
        }
    }

    // -------- head --------
    pool_kernel<<<ceil_div(BD * CHT, 256), 256, 0, stream>>>(XF, POOL);
    lin1_kernel<<<ceil_div(BD * 1024, 256), 256, 0, stream>>>(POOL, lin1_w, lin1_b, H1);
    bnstats_kernel<<<ceil_div(1024, 256), 256, 0, stream>>>(H1, MU, VAR);
    bnrelu_kernel<<<ceil_div(BD * 1024, 256), 256, 0, stream>>>(H1, MU, VAR, gamma, beta);
    lin2_kernel<<<ceil_div(BD * 40, 64), 64, 0, stream>>>(H1, lin2_w, lin2_b, LOG);
    logsoftmax_kernel<<<1, 32, 0, stream>>>(LOG, out);
}
